// RWKV6Attention_77017353552383
// MI455X (gfx1250) — matrix operand order repaired and checked by reading, not compiled, not run on hardware
//
#include <hip/hip_runtime.h>
#include <math.h>

// ---------------- problem constants (match reference) ----------------
#define Hc   1024
#define NHc  4
#define KDc  512
#define VDc  1024
#define DKc  128
#define DVc  256
#define Cc   32
#define PLRc 32
#define GLRc 64
#define Bc   2
#define Lc   2048
#define NTOK (Bc * Lc)      // 4096
#define NCHUNK (Lc / Cc)    // 64

// ---------------- WMMA types / helpers ----------------
typedef __attribute__((ext_vector_type(16))) __bf16 v16bf;
typedef __attribute__((ext_vector_type(8)))  float  v8f;

union FragU { v16bf v; uint4 q[2]; unsigned u[8]; unsigned short h[16]; };
union FragF { v8f v; float f[8]; };
union Pack8 { uint4 q; unsigned short h[8]; };

#define WMMA_BF16(a, b, c) \
  __builtin_amdgcn_wmma_f32_16x16x32_bf16(false, (a), false, (b), (short)0, (c), false, false)

__device__ __forceinline__ unsigned short f2bf(float x) {
  unsigned u = __float_as_uint(x);
  unsigned r = u + 0x7FFFu + ((u >> 16) & 1u);
  return (unsigned short)(r >> 16);
}
__device__ __forceinline__ float bf2f(unsigned short h) {
  return __uint_as_float(((unsigned)h) << 16);
}

// A fragment (16x32 bf16) from row-major matrix: lane r -> M=row0+r,
// K = (i<8 ? half*8+i : 16+half*8+(i-8))  -> two 16B contiguous loads
__device__ __forceinline__ v16bf gload_a(const unsigned short* A, int lda, int row0, int k0) {
  int lane = threadIdx.x & 31, hl = lane >> 4, r = lane & 15;
  const unsigned short* base = A + (size_t)(row0 + r) * lda + k0 + hl * 8;
  FragU f;
  f.q[0] = *(const uint4*)(base);
  f.q[1] = *(const uint4*)(base + 16);
  return f.v;
}

// B fragment (32x16 bf16) as W^T: B[k][n] = W[n][k]; lane r -> N=n0+r,
// K = (i<8 ? half*8+i : 16+half*8+(i-8))  -> two 16B loads (row clamp on edge)
__device__ __forceinline__ v16bf gload_bT(const unsigned short* W, int ldw, int n0, int k0, int N) {
  int lane = threadIdx.x & 31, hl = lane >> 4, r = lane & 15;
  int n = n0 + r; if (n >= N) n = N - 1;
  const unsigned short* base = W + (size_t)n * ldw + k0 + hl * 8;
  FragU f;
  f.q[0] = *(const uint4*)(base);
  f.q[1] = *(const uint4*)(base + 16);
  return f.v;
}

// Same patterns against LDS (compiler emits ds_load_b128 pairs)
__device__ __forceinline__ v16bf frag_a_lds(const unsigned short* P, int ld, int row0, int k0) {
  int lane = threadIdx.x & 31, hl = lane >> 4, r = lane & 15;
  const unsigned short* base = P + (size_t)(row0 + r) * ld + k0 + hl * 8;
  FragU f;
  f.q[0] = *(const uint4*)(base);
  f.q[1] = *(const uint4*)(base + 16);
  return f.v;
}
// B[k][n] = P[n*ld + k]  (transposed-stored source -> contiguous per lane)
__device__ __forceinline__ v16bf frag_bT_lds(const unsigned short* P, int ld, int n0, int k0) {
  int lane = threadIdx.x & 31, hl = lane >> 4, r = lane & 15;
  const unsigned short* base = P + (size_t)(n0 + r) * ld + k0 + hl * 8;
  FragU f;
  f.q[0] = *(const uint4*)(base);
  f.q[1] = *(const uint4*)(base + 16);
  return f.v;
}

// ---------------- small elementwise kernels ----------------
__global__ void cvt_f32_bf16(const float* __restrict__ in, unsigned short* __restrict__ out, int n) {
  int i = blockIdx.x * 256 + threadIdx.x;
  if (i < n) out[i] = f2bf(in[i]);
}

// a0 = x + (shift(x) - x) * mu0[k]   (token-shift lerp), bf16
__global__ void prep_a0(const float* __restrict__ x, const float* __restrict__ mu0,
                        unsigned short* __restrict__ a0) {
  int i = blockIdx.x * 256 + threadIdx.x;
  if (i >= NTOK * Hc) return;
  int row = i / Hc, k = i - row * Hc;
  int l = row & (Lc - 1);
  float xv = x[i];
  float sh = l ? x[i - Hc] : 0.f;
  a0[i] = f2bf(xv + (sh - xv) * mu0[k]);
}

// ---------------- generic bf16 WMMA GEMM: C = A[M,K] @ W[N,K]^T ----------------
// wave tile 32x64 (2 A-frags + 4 B-frags -> 8 WMMA per K-step), block tile 64x256
// EPI: 0 plain f32 | 1 tanh->bf16 | 2 mix: bf16(x + delta*(acc+bias)) | 3 f32(-exp(acc+bias)) | 4 f32 swish(acc+bias)
template <int EPI>
__global__ __launch_bounds__(256) void gemm_bf16(
    const unsigned short* __restrict__ A, int lda,
    const unsigned short* __restrict__ W, int ldw,
    int M, int N, int K,
    float* __restrict__ outf, unsigned short* __restrict__ outb, int ldo,
    const float* __restrict__ bias, const float* __restrict__ xsrc) {
  int wv = threadIdx.x >> 5;
  int lane = threadIdx.x & 31, hl = lane >> 4, r = lane & 15;
  int mBase = blockIdx.x * 64 + (wv & 1) * 32;
  int nBase = blockIdx.y * 256 + (wv >> 1) * 64;

  FragF acc[2][4];
#pragma unroll
  for (int a = 0; a < 2; ++a)
#pragma unroll
    for (int b = 0; b < 4; ++b)
#pragma unroll
      for (int j = 0; j < 8; ++j) acc[a][b].f[j] = 0.f;

  for (int k0 = 0; k0 < K; k0 += 32) {
    v16bf bfr[4];
#pragma unroll
    for (int nt = 0; nt < 4; ++nt) bfr[nt] = gload_bT(W, ldw, nBase + nt * 16, k0, N);
    v16bf a0 = gload_a(A, lda, mBase, k0);
    v16bf a1 = gload_a(A, lda, mBase + 16, k0);
#pragma unroll
    for (int nt = 0; nt < 4; ++nt) {
      acc[0][nt].v = WMMA_BF16(a0, bfr[nt], acc[0][nt].v);
      acc[1][nt].v = WMMA_BF16(a1, bfr[nt], acc[1][nt].v);
    }
  }

#pragma unroll
  for (int mt = 0; mt < 2; ++mt)
#pragma unroll
    for (int nt = 0; nt < 4; ++nt)
#pragma unroll
      for (int j = 0; j < 8; ++j) {
        int mrow = mBase + mt * 16 + j + hl * 8;
        int ncol = nBase + nt * 16 + r;
        if (ncol >= N) continue;
        float vv = acc[mt][nt].f[j];
        if (EPI == 0) {
          outf[(size_t)mrow * ldo + ncol] = vv;
        } else if (EPI == 1) {
          outb[(size_t)mrow * ldo + ncol] = f2bf(tanhf(vv));
        } else if (EPI == 2) {
          vv += bias[ncol];
          size_t xi = (size_t)mrow * Hc + ncol;
          float xv = xsrc[xi];
          int l = mrow & (Lc - 1);
          float sh = l ? xsrc[xi - Hc] : 0.f;
          outb[(size_t)mrow * ldo + ncol] = f2bf(xv + (sh - xv) * vv);
        } else if (EPI == 3) {
          vv += bias[ncol];
          outf[(size_t)mrow * ldo + ncol] = -__expf(vv);
        } else { // 4: swish gate value g*sigmoid(g)
          vv += bias[ncol];
          outf[(size_t)mrow * ldo + ncol] = vv / (1.f + __expf(-vv));
        }
      }
}

// ---------------- chunked RWKV6 recurrence ----------------
// grid = B*NH*4 workgroups (dv split into 4 slices of 64), block = 128 (4 waves)
// state S[128, 64-slice] fp32 in register fragments + transposed bf16 LDS mirror
__global__ __launch_bounds__(128) void rwkv6_chunk_kernel(
    const float* __restrict__ q, const float* __restrict__ k,
    const float* __restrict__ w, const float* __restrict__ vsrc,
    const float* __restrict__ bonus, float* __restrict__ o) {
  __shared__ float wc_s[32 * 128];
  __shared__ float dec_s[128];
  __shared__ unsigned short qe_s[32 * 128];   // [i][d]  (A operand, row-contig)
  __shared__ unsigned short kd_s[32 * 128];   // [j][d]  (B^T source, contig)
  __shared__ unsigned short kw_sT[128 * 32];  // [d][c]  (A operand for kw^T @ v)
  __shared__ unsigned short v_sT[64 * 32];    // [p][i]  (B^T source, contig)
  __shared__ unsigned short S_sT[64 * 128];   // [p][d]  (B^T source, contig)
  __shared__ unsigned short attn_s[32 * 32];  // [i][j]  (A operand, row-contig)
  __shared__ float dsum[32];

  int wg = blockIdx.x;
  int b = wg >> 4;
  int h = (wg >> 2) & 3;
  int vs = wg & 3;
  int tid = threadIdx.x;
  int wv = tid >> 5, lane = tid & 31, hl = lane >> 4, r = lane & 15;
  int cw = wv * 16;

  FragF Sacc[8];
#pragma unroll
  for (int t = 0; t < 8; ++t)
#pragma unroll
    for (int j = 0; j < 8; ++j) Sacc[t].f[j] = 0.f;
  for (int t = tid; t < 64 * 128; t += 128) S_sT[t] = 0;

  const size_t qkwBase = ((size_t)b * Lc) * KDc + (size_t)h * DKc;
  const size_t vBase = ((size_t)b * Lc) * VDc + (size_t)h * DVc + (size_t)vs * 64;

  for (int ci = 0; ci < NCHUNK; ++ci) {
    __syncthreads();
    if (tid < 32) dsum[tid] = 0.f;
    if (tid < 128) {  // per-d cumsum of decay
      int d = tid;
      float acc = 0.f;
#pragma unroll 4
      for (int i = 0; i < 32; ++i) {
        acc += w[qkwBase + (size_t)(ci * 32 + i) * KDc + d];
        wc_s[i * 128 + d] = acc;
      }
      dec_s[d] = __expf(acc);
      if (ci + 1 < NCHUNK) {  // prefetch next chunk (global_prefetch_b8)
        size_t nb = qkwBase + (size_t)((ci + 1) * 32) * KDc + d;
        __builtin_prefetch(&q[nb], 0, 1);
        __builtin_prefetch(&k[nb], 0, 1);
        __builtin_prefetch(&w[nb], 0, 1);
      }
    }
    __syncthreads();
    if (tid < 128) {  // build bf16 operands + bonus diag partials
      int d = tid;
      float wcend = wc_s[31 * 128 + d];
      float u = bonus[h * DKc + d];
#pragma unroll 4
      for (int i = 0; i < 32; ++i) {
        size_t gi = qkwBase + (size_t)(ci * 32 + i) * KDc + d;
        float wi = w[gi], ki = k[gi], qi = q[gi];
        float wci = wc_s[i * 128 + d];
        kw_sT[d * 32 + i] = f2bf(ki * __expf(wcend - wci));
        kd_s[i * 128 + d] = f2bf(ki * __expf(-wci));
        qe_s[i * 128 + d] = f2bf(qi * __expf(wci - wi));
        atomicAdd(&dsum[i], qi * u * ki);
      }
    }
    for (int t = tid; t < 32 * 64; t += 128) {  // v slice, coalesced read, transposed store
      int row = t >> 6, col = t & 63;
      v_sT[col * 32 + row] = f2bf(vsrc[vBase + (size_t)(ci * 32 + row) * VDc + col]);
    }
    __syncthreads();

    // o_inter = qe @ S_old  (this wave's 16-column strip)
    FragF oacc[2];
#pragma unroll
    for (int mt = 0; mt < 2; ++mt)
#pragma unroll
      for (int j = 0; j < 8; ++j) oacc[mt].f[j] = 0.f;
#pragma unroll
    for (int kk = 0; kk < 4; ++kk) {
      v16bf bf = frag_bT_lds(S_sT, 128, cw, kk * 32);
#pragma unroll
      for (int mt = 0; mt < 2; ++mt) {
        v16bf af = frag_a_lds(qe_s, 128, mt * 16, kk * 32);
        oacc[mt].v = WMMA_BF16(af, bf, oacc[mt].v);
      }
    }

    // attn = (qe @ kd^T), one 16x16 tile per wave, strict-lower mask
    {
      int mt = wv & 1, nt = wv >> 1;
      FragF at;
#pragma unroll
      for (int j = 0; j < 8; ++j) at.f[j] = 0.f;
#pragma unroll
      for (int kk = 0; kk < 4; ++kk) {
        v16bf af = frag_a_lds(qe_s, 128, mt * 16, kk * 32);
        v16bf bf = frag_bT_lds(kd_s, 128, nt * 16, kk * 32);
        at.v = WMMA_BF16(af, bf, at.v);
      }
#pragma unroll
      for (int j = 0; j < 8; ++j) {
        int i = mt * 16 + j + hl * 8;
        int jj = nt * 16 + r;
        attn_s[i * 32 + jj] = f2bf((jj < i) ? at.f[j] : 0.f);
      }
    }
    __syncthreads();

    // o_intra += attn @ v
    {
      v16bf bf = frag_bT_lds(v_sT, 32, cw, 0);
#pragma unroll
      for (int mt = 0; mt < 2; ++mt) {
        v16bf af = frag_a_lds(attn_s, 32, mt * 16, 0);
        oacc[mt].v = WMMA_BF16(af, bf, oacc[mt].v);
      }
    }

    // store o chunk (+ bonus diagonal term)
#pragma unroll
    for (int mt = 0; mt < 2; ++mt)
#pragma unroll
      for (int j = 0; j < 8; ++j) {
        int i = mt * 16 + j + hl * 8;
        int p = cw + r;
        float val = oacc[mt].f[j] + dsum[i] * bf2f(v_sT[p * 32 + i]);
        o[vBase + (size_t)(ci * 32 + i) * VDc + p] = val;
      }

    // state update: S = S*dec + kw^T @ v  (each wave its own strip -> no race)
    {
      v16bf bf = frag_bT_lds(v_sT, 32, cw, 0);
#pragma unroll
      for (int mt8 = 0; mt8 < 8; ++mt8) {
        v16bf af = frag_a_lds(kw_sT, 32, mt8 * 16, 0);
        FragF ua;
#pragma unroll
        for (int j = 0; j < 8; ++j) ua.f[j] = 0.f;
        ua.v = WMMA_BF16(af, bf, ua.v);
        Pack8 pk;
#pragma unroll
        for (int j = 0; j < 8; ++j) {
          int d = mt8 * 16 + j + hl * 8;
          float ns = Sacc[mt8].f[j] * dec_s[d] + ua.f[j];
          Sacc[mt8].f[j] = ns;
          pk.h[j] = f2bf(ns);
        }
        // contiguous 8 x bf16 mirror writeback (16B aligned)
        *(uint4*)(&S_sT[(cw + r) * 128 + mt8 * 16 + hl * 8]) = pk.q;
      }
    }
  }
}

// ---------------- LayerNorm + swish gate -> bf16 ----------------
__global__ __launch_bounds__(256) void lngate_kernel(
    const float* __restrict__ o, const float* __restrict__ gg,
    const float* __restrict__ ln_g, const float* __restrict__ ln_b,
    unsigned short* __restrict__ out) {
  __shared__ float sb[256];
  int row = blockIdx.x, t = threadIdx.x;
  const float* orow = o + (size_t)row * VDc;
  float vals[4], s = 0.f;
#pragma unroll
  for (int j = 0; j < 4; ++j) { vals[j] = orow[t + j * 256]; s += vals[j]; }
  sb[t] = s; __syncthreads();
  for (int st = 128; st > 0; st >>= 1) { if (t < st) sb[t] += sb[t + st]; __syncthreads(); }
  float mean = sb[0] * (1.f / VDc);
  __syncthreads();
  float s2 = 0.f;
#pragma unroll
  for (int j = 0; j < 4; ++j) { float d = vals[j] - mean; s2 += d * d; }
  sb[t] = s2; __syncthreads();
  for (int st = 128; st > 0; st >>= 1) { if (t < st) sb[t] += sb[t + st]; __syncthreads(); }
  float rstd = rsqrtf(sb[0] * (1.f / VDc) + 1e-5f);
#pragma unroll
  for (int j = 0; j < 4; ++j) {
    int col = t + j * 256;
    float y = (vals[j] - mean) * rstd * ln_g[col] + ln_b[col];
    y *= gg[(size_t)row * VDc + col];
    out[(size_t)row * VDc + col] = f2bf(y);
  }
}

// ---------------- host launch ----------------
extern "C" void kernel_launch(void* const* d_in, const int* in_sizes, int n_in,
                              void* d_out, int out_size, void* d_ws, size_t ws_size,
                              hipStream_t stream) {
  const float* x      = (const float*)d_in[0];
  const float* mu0    = (const float*)d_in[1];
  const float* Wx0    = (const float*)d_in[2];
  const float* Wx2    = (const float*)d_in[3];
  const float* x_bias = (const float*)d_in[4];
  const float* rW     = (const float*)d_in[5];
  const float* wA     = (const float*)d_in[6];
  const float* wB     = (const float*)d_in[7];
  const float* wb     = (const float*)d_in[8];
  const float* kW     = (const float*)d_in[9];
  const float* vW     = (const float*)d_in[10];
  const float* gA     = (const float*)d_in[11];
  const float* gB     = (const float*)d_in[12];
  const float* gb     = (const float*)d_in[13];
  const float* bonus  = (const float*)d_in[14];
  const float* ln_g   = (const float*)d_in[15];
  const float* ln_b   = (const float*)d_in[16];
  const float* oW     = (const float*)d_in[17];
  float* out = (float*)d_out;

  char* wsp = (char*)d_ws;
  size_t off = 0;
  auto take = [&](size_t bytes) -> void* {
    void* p = wsp + off;
    off = (off + bytes + 255) & ~(size_t)255;
    return p;
  };
  unsigned short* Wx0_b = (unsigned short*)take((size_t)160 * 1024 * 2);
  unsigned short* Wx2_b = (unsigned short*)take((size_t)1024 * 160 * 2);
  unsigned short* rW_b  = (unsigned short*)take((size_t)512 * 1024 * 2);
  unsigned short* wA_b  = (unsigned short*)take((size_t)64 * 1024 * 2);
  unsigned short* wB_b  = (unsigned short*)take((size_t)512 * 64 * 2);
  unsigned short* kW_b  = (unsigned short*)take((size_t)512 * 1024 * 2);
  unsigned short* vW_b  = (unsigned short*)take((size_t)1024 * 1024 * 2);
  unsigned short* gA_b  = (unsigned short*)take((size_t)64 * 1024 * 2);
  unsigned short* gB_b  = (unsigned short*)take((size_t)1024 * 64 * 2);
  unsigned short* oW_b  = (unsigned short*)take((size_t)1024 * 1024 * 2);
  unsigned short* a0_b  = (unsigned short*)take((size_t)NTOK * 1024 * 2);
  unsigned short* xx_b  = (unsigned short*)take((size_t)NTOK * 160 * 2);
  unsigned short* inp_b = (unsigned short*)take((size_t)NTOK * 1024 * 2);
  unsigned short* t_b   = (unsigned short*)take((size_t)NTOK * 64 * 2);
  float* r_f  = (float*)take((size_t)NTOK * 512 * 4);
  float* k_f  = (float*)take((size_t)NTOK * 512 * 4);
  float* w_f  = (float*)take((size_t)NTOK * 512 * 4);
  float* v_f  = (float*)take((size_t)NTOK * 1024 * 4);
  float* gg_f = (float*)take((size_t)NTOK * 1024 * 4);
  float* o_f  = (float*)take((size_t)NTOK * 1024 * 4);
  unsigned short* o2_b = (unsigned short*)take((size_t)NTOK * 1024 * 2);
  (void)ws_size; (void)n_in; (void)in_sizes; (void)out_size;

  auto cvt = [&](const float* src, unsigned short* dst, int n) {
    cvt_f32_bf16<<<(n + 255) / 256, 256, 0, stream>>>(src, dst, n);
  };
  cvt(Wx0, Wx0_b, 160 * 1024);
  cvt(Wx2, Wx2_b, 1024 * 160);
  cvt(rW, rW_b, 512 * 1024);
  cvt(wA, wA_b, 64 * 1024);
  cvt(wB, wB_b, 512 * 64);
  cvt(kW, kW_b, 512 * 1024);
  cvt(vW, vW_b, 1024 * 1024);
  cvt(gA, gA_b, 64 * 1024);
  cvt(gB, gB_b, 1024 * 64);
  cvt(oW, oW_b, 1024 * 1024);

  prep_a0<<<(NTOK * Hc + 255) / 256, 256, 0, stream>>>(x, mu0, a0_b);

  dim3 blk(256);
  auto grid = [](int M, int N) { return dim3(M / 64, (N + 255) / 256); };

  // xx = tanh(a0 @ Wx0^T) -> bf16 [4096,160]
  gemm_bf16<1><<<grid(NTOK, 160), blk, 0, stream>>>(
      a0_b, 1024, Wx0_b, 1024, NTOK, 160, 1024,
      nullptr, xx_b, 160, nullptr, nullptr);

  for (int i = 0; i < 5; ++i) {
    // mix_i -> lerped input (bf16), fused token-shift epilogue
    gemm_bf16<2><<<grid(NTOK, 1024), blk, 0, stream>>>(
        xx_b + i * 32, 160, Wx2_b + i * 32, 160, NTOK, 1024, 32,
        nullptr, inp_b, 1024, x_bias + i * Hc, x);
    if (i == 0) {       // r
      gemm_bf16<0><<<grid(NTOK, 512), blk, 0, stream>>>(
          inp_b, 1024, rW_b, 1024, NTOK, 512, 1024,
          r_f, nullptr, 512, nullptr, nullptr);
    } else if (i == 1) { // w LoRA -> -exp
      gemm_bf16<1><<<grid(NTOK, 64), blk, 0, stream>>>(
          inp_b, 1024, wA_b, 1024, NTOK, 64, 1024,
          nullptr, t_b, 64, nullptr, nullptr);
      gemm_bf16<3><<<grid(NTOK, 512), blk, 0, stream>>>(
          t_b, 64, wB_b, 64, NTOK, 512, 64,
          w_f, nullptr, 512, wb, nullptr);
    } else if (i == 2) { // k
      gemm_bf16<0><<<grid(NTOK, 512), blk, 0, stream>>>(
          inp_b, 1024, kW_b, 1024, NTOK, 512, 1024,
          k_f, nullptr, 512, nullptr, nullptr);
    } else if (i == 3) { // v
      gemm_bf16<0><<<grid(NTOK, 1024), blk, 0, stream>>>(
          inp_b, 1024, vW_b, 1024, NTOK, 1024, 1024,
          v_f, nullptr, 1024, nullptr, nullptr);
    } else {             // g LoRA -> swish gate
      gemm_bf16<1><<<grid(NTOK, 64), blk, 0, stream>>>(
          inp_b, 1024, gA_b, 1024, NTOK, 64, 1024,
          nullptr, t_b, 64, nullptr, nullptr);
      gemm_bf16<4><<<grid(NTOK, 1024), blk, 0, stream>>>(
          t_b, 64, gB_b, 64, NTOK, 1024, 64,
          gg_f, nullptr, 1024, gb, nullptr);
    }
  }

  // chunked recurrence
  rwkv6_chunk_kernel<<<Bc * NHc * 4, 128, 0, stream>>>(r_f, k_f, w_f, v_f, bonus, o_f);

  // LayerNorm + gate
  lngate_kernel<<<NTOK, 256, 0, stream>>>(o_f, gg_f, ln_g, ln_b, o2_b);

  // final projection -> fp32 output
  gemm_bf16<0><<<grid(NTOK, 1024), blk, 0, stream>>>(
      o2_b, 1024, oW_b, 1024, NTOK, 1024, 1024,
      out, nullptr, 1024, nullptr, nullptr);
}
